// RNN_62345745269268
// MI455X (gfx1250) — matrix operand order repaired and checked by reading, not compiled, not run on hardware
//
#include <hip/hip_runtime.h>
#include <hip/hip_bf16.h>

// RNN: T=512 steps of h = sigmoid(h @ W_h^T + x_t @ W_in^T + b_h + b_in), B=256, H=D=1024.
// Strategy: fuse W_h|W_in into one K=2048 bf16 operand (packed once, WMMA-fragment order),
// one bf16 WMMA GEMM kernel per timestep (graph-replayed), 2x2 register tiling (32x32 per
// wave -> 4 v_wmma per K-tile, halves L2 traffic per FLOP), fused sigmoid epilogue writing
// fp32 + bf16 hidden state (ping-pong, L2 resident). Final log-softmax head.

typedef __attribute__((ext_vector_type(16))) __bf16        v16bf;
typedef __attribute__((ext_vector_type(8)))  float         v8f;
typedef __attribute__((ext_vector_type(8)))  unsigned int  v8u;
typedef __attribute__((ext_vector_type(4)))  unsigned int  v4u;
typedef __attribute__((ext_vector_type(4)))  float         v4f;

#define T_STEPS 512
#define BATCH   256
#define HDIM    1024

__device__ __forceinline__ unsigned int bf16_rne(float f) {
    unsigned int u = __builtin_bit_cast(unsigned int, f);
    u += 0x7FFFu + ((u >> 16) & 1u);   // round-to-nearest-even (finite inputs)
    return u >> 16;
}
__device__ __forceinline__ unsigned int pack_bf16x2(float lo, float hi) {
    return bf16_rne(lo) | (bf16_rne(hi) << 16);
}
__device__ __forceinline__ v16bf frag_from(v4u a, v4u b) {
    v8u u = {a[0], a[1], a[2], a[3], b[0], b[1], b[2], b[3]};
    return __builtin_bit_cast(v16bf, u);
}

// ---------------------------------------------------------------------------
// Pack [W_h ; W_in] (each [1024,1024] row-major fp32) into bf16 B-matrix
// fragments: Wc[((ntile*64 + ktile)*32 + lane)*8 + v] holds the dword for
// VGPR v of lane `lane` of K-tile `ktile`, N-tile `ntile`.
// B 32x16 bf16 layout: lane = N + 16*khalf, dword v = K pair (8*khalf+2v, +1) for v<4, (16+8*khalf+2(v-4), +1) for v>=4.
// B[k][n] = W[n][k]  (so D = A x B computes rows @ W^T).
// ---------------------------------------------------------------------------
__global__ __launch_bounds__(256) void pack_weights(
    const float* __restrict__ Wh, const float* __restrict__ Win,
    unsigned int* __restrict__ Wc)
{
    unsigned int idx  = blockIdx.x * 256u + threadIdx.x;   // 0 .. 2^20-1 dwords
    unsigned int v    = idx & 7u;
    unsigned int lane = (idx >> 3) & 31u;
    unsigned int kt   = (idx >> 8) & 63u;
    unsigned int nt   = idx >> 14;
    unsigned int n    = nt * 16u + (lane & 15u);
    unsigned int K0   = kt * 32u + (v >> 2) * 16u + (lane >> 4) * 8u + 2u * (v & 3u);
    float w0, w1;
    if (K0 < 1024u) {
        w0 = Wh[n * 1024u + K0];
        w1 = Wh[n * 1024u + K0 + 1u];
    } else {
        w0 = Win[n * 1024u + (K0 - 1024u)];
        w1 = Win[n * 1024u + (K0 - 1024u) + 1u];
    }
    Wc[idx] = pack_bf16x2(w0, w1);
}

// fp32 -> packed bf16 pairs (for the initial hidden state)
__global__ __launch_bounds__(256) void f32_to_bf16_pairs(
    const float* __restrict__ src, unsigned int* __restrict__ dst, int n_pairs)
{
    int i = blockIdx.x * 256 + threadIdx.x;
    if (i < n_pairs) dst[i] = pack_bf16x2(src[2 * i], src[2 * i + 1]);
}

// ---------------------------------------------------------------------------
// One RNN timestep: h_new = sigmoid([h_bf16 | x_t] @ [W_h;W_in]^T + b_h + b_in)
// 256 waves, each computing a 32x32 output patch (2x2 WMMA tiles, 4 v8f accs).
// K = 2048 -> 64 K-tiles, 4 WMMAs each.
// A 16x32 bf16 layout: lane = M + 16*khalf; dwords v0..3 -> K pairs
// (kh*8 + 2v, +1), v4..7 -> (16 + kh*8 + 2(v-4), +1).
// ---------------------------------------------------------------------------
__global__ __launch_bounds__(256) void rnn_step(
    const unsigned int* __restrict__ Wc,
    const unsigned int* __restrict__ hbf_in,   // bf16 pairs, [256][512] dwords
    const float*        __restrict__ x_t,      // [256][1024] fp32
    const float*        __restrict__ b_h,
    const float*        __restrict__ b_in,
    float*              __restrict__ h_out,    // [256][1024] fp32
    unsigned short*     __restrict__ hbf_out)  // [256][1024] bf16
{
    const unsigned int wave = threadIdx.x >> 5;
    const unsigned int lane = threadIdx.x & 31u;
    const unsigned int tile = blockIdx.x * 8u + wave;   // 0..255
    const unsigned int mb   = tile & 7u;                // 8 M super-tiles (32 rows)
    const unsigned int nb   = tile >> 3;                // 32 N super-tiles (32 cols)
    const unsigned int ln   = lane & 15u;
    const unsigned int kh   = lane >> 4;

    const unsigned int row0  = mb * 32u + ln;           // A tile 0 rows
    const unsigned int row1  = row0 + 16u;              // A tile 1 rows
    const unsigned int ncol0 = nb * 32u + ln;           // B/D tile 0 column
    const unsigned int ncol1 = ncol0 + 16u;             // B/D tile 1 column

    v8f acc00 = {0.f,0.f,0.f,0.f,0.f,0.f,0.f,0.f};
    v8f acc01 = acc00, acc10 = acc00, acc11 = acc00;

    // B fragment base: nt stride in Wc = 64*32*8 = 16384 dwords; kt stride = 256.
    const unsigned int* bptr  = Wc + ((nb * 2u) * 16384u + lane * 8u);
    const unsigned int* hrow0 = hbf_in + row0 * 512u;   // bf16 pair dwords
    const unsigned int* hrow1 = hbf_in + row1 * 512u;

    // K = 0..1023 : A from hidden state (already bf16, fragment-friendly pairs)
    for (unsigned int kt = 0; kt < 32u; ++kt) {
        v4u b0a = *(const v4u*)(bptr);
        v4u b0b = *(const v4u*)(bptr + 4);
        v4u b1a = *(const v4u*)(bptr + 16384);
        v4u b1b = *(const v4u*)(bptr + 16388);
        __builtin_prefetch(bptr + 256, 0, 0);           // global_prefetch_b8
        __builtin_prefetch(bptr + 16384 + 256, 0, 0);
        bptr += 256;

        const unsigned int* ap0 = hrow0 + kt * 16u + kh * 4u;
        const unsigned int* ap1 = hrow1 + kt * 16u + kh * 4u;
        v4u a0a = *(const v4u*)(ap0);
        v4u a0b = *(const v4u*)(ap0 + 8);
        v4u a1a = *(const v4u*)(ap1);
        v4u a1b = *(const v4u*)(ap1 + 8);

        v16bf A0 = frag_from(a0a, a0b);
        v16bf A1 = frag_from(a1a, a1b);
        v16bf B0 = frag_from(b0a, b0b);
        v16bf B1 = frag_from(b1a, b1b);
        acc00 = __builtin_amdgcn_wmma_f32_16x16x32_bf16(false, A0, false, B0, (short)0, acc00, false, false);
        acc01 = __builtin_amdgcn_wmma_f32_16x16x32_bf16(false, A0, false, B1, (short)0, acc01, false, false);
        acc10 = __builtin_amdgcn_wmma_f32_16x16x32_bf16(false, A1, false, B0, (short)0, acc10, false, false);
        acc11 = __builtin_amdgcn_wmma_f32_16x16x32_bf16(false, A1, false, B1, (short)0, acc11, false, false);
    }

    // K = 1024..2047 : A from x_t (fp32 -> bf16 in registers)
    const float* xrow0 = x_t + row0 * 1024u;
    const float* xrow1 = x_t + row1 * 1024u;
    for (unsigned int kt = 32u; kt < 64u; ++kt) {
        v4u b0a = *(const v4u*)(bptr);
        v4u b0b = *(const v4u*)(bptr + 4);
        v4u b1a = *(const v4u*)(bptr + 16384);
        v4u b1b = *(const v4u*)(bptr + 16388);
        __builtin_prefetch(bptr + 256, 0, 0);
        __builtin_prefetch(bptr + 16384 + 256, 0, 0);
        bptr += 256;

        const unsigned int xoff = (kt - 32u) * 32u + kh * 8u;
        const float* xp0 = xrow0 + xoff;
        const float* xp1 = xrow1 + xoff;
        v4f f00 = *(const v4f*)(xp0);
        v4f f01 = *(const v4f*)(xp0 + 4);
        v4f f02 = *(const v4f*)(xp0 + 16);
        v4f f03 = *(const v4f*)(xp0 + 20);
        v4f f10 = *(const v4f*)(xp1);
        v4f f11 = *(const v4f*)(xp1 + 4);
        v4f f12 = *(const v4f*)(xp1 + 16);
        v4f f13 = *(const v4f*)(xp1 + 20);

        v8u a0u = {pack_bf16x2(f00[0], f00[1]), pack_bf16x2(f00[2], f00[3]),
                   pack_bf16x2(f01[0], f01[1]), pack_bf16x2(f01[2], f01[3]),
                   pack_bf16x2(f02[0], f02[1]), pack_bf16x2(f02[2], f02[3]),
                   pack_bf16x2(f03[0], f03[1]), pack_bf16x2(f03[2], f03[3])};
        v8u a1u = {pack_bf16x2(f10[0], f10[1]), pack_bf16x2(f10[2], f10[3]),
                   pack_bf16x2(f11[0], f11[1]), pack_bf16x2(f11[2], f11[3]),
                   pack_bf16x2(f12[0], f12[1]), pack_bf16x2(f12[2], f12[3]),
                   pack_bf16x2(f13[0], f13[1]), pack_bf16x2(f13[2], f13[3])};
        v16bf A0 = __builtin_bit_cast(v16bf, a0u);
        v16bf A1 = __builtin_bit_cast(v16bf, a1u);
        v16bf B0 = frag_from(b0a, b0b);
        v16bf B1 = frag_from(b1a, b1b);
        acc00 = __builtin_amdgcn_wmma_f32_16x16x32_bf16(false, A0, false, B0, (short)0, acc00, false, false);
        acc01 = __builtin_amdgcn_wmma_f32_16x16x32_bf16(false, A0, false, B1, (short)0, acc01, false, false);
        acc10 = __builtin_amdgcn_wmma_f32_16x16x32_bf16(false, A1, false, B0, (short)0, acc10, false, false);
        acc11 = __builtin_amdgcn_wmma_f32_16x16x32_bf16(false, A1, false, B1, (short)0, acc11, false, false);
    }

    // Epilogue: bias + sigmoid, store fp32 and bf16 hidden.
    // D layout: VGPR r -> M = r + 8*kh, N = ncol.
    const float bias0 = b_h[ncol0] + b_in[ncol0];
    const float bias1 = b_h[ncol1] + b_in[ncol1];
#pragma unroll
    for (int r = 0; r < 8; ++r) {
        const unsigned int g0 = mb * 32u + (unsigned int)r + kh * 8u; // tile rows 0..15
        const unsigned int g1 = g0 + 16u;                             // tile rows 16..31
        float v00 = 1.0f / (1.0f + __expf(-(acc00[r] + bias0)));
        float v01 = 1.0f / (1.0f + __expf(-(acc01[r] + bias1)));
        float v10 = 1.0f / (1.0f + __expf(-(acc10[r] + bias0)));
        float v11 = 1.0f / (1.0f + __expf(-(acc11[r] + bias1)));
        h_out[g0 * 1024u + ncol0] = v00;
        h_out[g0 * 1024u + ncol1] = v01;
        h_out[g1 * 1024u + ncol0] = v10;
        h_out[g1 * 1024u + ncol1] = v11;
        hbf_out[g0 * 1024u + ncol0] = (unsigned short)bf16_rne(v00);
        hbf_out[g0 * 1024u + ncol1] = (unsigned short)bf16_rne(v01);
        hbf_out[g1 * 1024u + ncol0] = (unsigned short)bf16_rne(v10);
        hbf_out[g1 * 1024u + ncol1] = (unsigned short)bf16_rne(v11);
    }
}

// ---------------------------------------------------------------------------
// out[b, :] = log_softmax(h[b,:] @ W_o^T + b_o), O = 5. One block per batch row.
// ---------------------------------------------------------------------------
__global__ __launch_bounds__(256) void finalize_logits(
    const float* __restrict__ h, const float* __restrict__ Wo,
    const float* __restrict__ bo, float* __restrict__ out)
{
    __shared__ float red[5][256];
    const int b = blockIdx.x;
    const int t = threadIdx.x;
    float p[5] = {0.f, 0.f, 0.f, 0.f, 0.f};
    for (int k = t; k < HDIM; k += 256) {
        float hv = h[b * HDIM + k];
#pragma unroll
        for (int o = 0; o < 5; ++o) p[o] += hv * Wo[o * HDIM + k];
    }
#pragma unroll
    for (int o = 0; o < 5; ++o) red[o][t] = p[o];
    __syncthreads();
    for (int s = 128; s > 0; s >>= 1) {
        if (t < s) {
#pragma unroll
            for (int o = 0; o < 5; ++o) red[o][t] += red[o][t + s];
        }
        __syncthreads();
    }
    if (t == 0) {
        float logit[5], m = -INFINITY;
#pragma unroll
        for (int o = 0; o < 5; ++o) { logit[o] = red[o][0] + bo[o]; m = fmaxf(m, logit[o]); }
        float s = 0.f;
#pragma unroll
        for (int o = 0; o < 5; ++o) s += __expf(logit[o] - m);
        float lse = m + __logf(s);
#pragma unroll
        for (int o = 0; o < 5; ++o) out[b * 5 + o] = logit[o] - lse;
    }
}

// hidden -> d_out[1280 ..], state passthrough -> d_out[1280+262144 ..]
__global__ __launch_bounds__(256) void finalize_copy(
    const float* __restrict__ h, const float* __restrict__ state,
    float* __restrict__ out)
{
    int i = blockIdx.x * 256 + threadIdx.x;
    if (i < BATCH * HDIM) {
        out[1280 + i]                = h[i];
        out[1280 + BATCH * HDIM + i] = state[i];
    }
}

extern "C" void kernel_launch(void* const* d_in, const int* in_sizes, int n_in,
                              void* d_out, int out_size, void* d_ws, size_t ws_size,
                              hipStream_t stream) {
    (void)in_sizes; (void)n_in; (void)out_size; (void)ws_size;
    const float* x   = (const float*)d_in[0];  // [512,256,1024]
    const float* h0  = (const float*)d_in[1];  // [256,1024]
    const float* st  = (const float*)d_in[2];  // [256,1024]
    const float* Win = (const float*)d_in[3];  // [1024,1024]
    const float* bin = (const float*)d_in[4];  // [1024]
    const float* Wh  = (const float*)d_in[5];  // [1024,1024]
    const float* bh  = (const float*)d_in[6];  // [1024]
    const float* Wo  = (const float*)d_in[7];  // [5,1024]
    const float* bo  = (const float*)d_in[8];  // [5]
    float* out = (float*)d_out;

    char* ws = (char*)d_ws;
    unsigned int* Wc   = (unsigned int*)(ws);                                  // 4 MB
    unsigned int* hbf0 = (unsigned int*)(ws + (4u << 20));                     // 512 KB
    unsigned int* hbf1 = (unsigned int*)(ws + (4u << 20) + (512u << 10));      // 512 KB
    float*        hf32 = (float*)(ws + (5u << 20));                            // 1 MB

    pack_weights<<<4096, 256, 0, stream>>>(Wh, Win, Wc);
    f32_to_bf16_pairs<<<512, 256, 0, stream>>>(h0, hbf0, (BATCH * HDIM) / 2);

    for (int t = 0; t < T_STEPS; ++t) {
        const unsigned int* hin = (t & 1) ? hbf1 : hbf0;
        unsigned short* hout = (unsigned short*)((t & 1) ? hbf0 : hbf1);
        rnn_step<<<32, 256, 0, stream>>>(Wc, hin,
                                         x + (size_t)t * BATCH * HDIM,
                                         bh, bin, hf32, hout);
    }

    finalize_logits<<<BATCH, 256, 0, stream>>>(hf32, Wo, bo, out);
    finalize_copy<<<(BATCH * HDIM + 255) / 256, 256, 0, stream>>>(hf32, st, out);
}
